// AutoQVLALinearW4A4_56581899157843
// MI455X (gfx1250) — 8-bit matrix operand order repaired and checked by reading, compiled, not run on hardware
//
#include <hip/hip_runtime.h>
#include <hip/hip_fp16.h>

typedef int v8i __attribute__((ext_vector_type(8)));

#define K_DIM 4096
#define N_DIM 4096
#define KP_DIM (K_DIM / 2)

__global__ __launch_bounds__(256)
void quant_act_kernel(const __half* __restrict__ x,
                      signed char* __restrict__ aq,
                      float* __restrict__ ascale) {
    __shared__ float red[256];
    const int row = blockIdx.x;
    const int t   = threadIdx.x;

    const __half2* xr2 = (const __half2*)(x + (size_t)row * K_DIM + t * 16);

    float v[16];
    float mx = 0.f;
#pragma unroll
    for (int i = 0; i < 8; ++i) {
        float2 f = __half22float2(xr2[i]);
        v[2 * i]     = f.x;
        v[2 * i + 1] = f.y;
        mx = fmaxf(mx, fmaxf(fabsf(f.x), fabsf(f.y)));
    }

    red[t] = mx;
    __syncthreads();
#pragma unroll
    for (int s = 128; s > 0; s >>= 1) {
        if (t < s) red[t] = fmaxf(red[t], red[t + s]);
        __syncthreads();
    }
    const float scale = fmaxf(red[0], 1e-6f) / 7.0f;
    const float inv   = 1.0f / scale;

    unsigned int o[4];
#pragma unroll
    for (int i = 0; i < 4; ++i) {
        unsigned int w = 0;
#pragma unroll
        for (int j = 0; j < 4; ++j) {
            float q = rintf(v[4 * i + j] * inv);
            q = fminf(fmaxf(q, -8.0f), 7.0f);
            int qi = (int)q;
            w |= ((unsigned int)(qi & 0xFF)) << (8 * j);
        }
        o[i] = w;
    }
    uint4 pack;
    pack.x = o[0]; pack.y = o[1]; pack.z = o[2]; pack.w = o[3];
    *(uint4*)(aq + (size_t)row * K_DIM + t * 16) = pack;

    if (t == 0) ascale[row] = scale;
}

__global__ __launch_bounds__(256)
void unpack_w_kernel(const int* __restrict__ qw,
                     signed char* __restrict__ w8,
                     int total  ) {
    const int i = blockIdx.x * blockDim.x + threadIdx.x;
    if (i >= total) return;

    const int4* p = (const int4*)(qw + (size_t)i * 8);
    int4 a = p[0];
    int4 b = p[1];
    int src[8] = {a.x, a.y, a.z, a.w, b.x, b.y, b.z, b.w};

    unsigned int o[4];
#pragma unroll
    for (int j = 0; j < 4; ++j) {
        int v0 = src[2 * j];
        int v1 = src[2 * j + 1];
        int l0 = ((v0 & 15) ^ 8) - 8;
        int h0 = (((v0 >> 4) & 15) ^ 8) - 8;
        int l1 = ((v1 & 15) ^ 8) - 8;
        int h1 = (((v1 >> 4) & 15) ^ 8) - 8;
        o[j] = ((unsigned int)(l0 & 0xFF)) |
               (((unsigned int)(h0 & 0xFF)) << 8) |
               (((unsigned int)(l1 & 0xFF)) << 16) |
               (((unsigned int)(h1 & 0xFF)) << 24);
    }
    uint4 pack;
    pack.x = o[0]; pack.y = o[1]; pack.z = o[2]; pack.w = o[3];
    ((uint4*)w8)[i] = pack;
}

__global__ __launch_bounds__(256)
void gemm_w4a4_kernel(const signed char* __restrict__ aq,
                      const float* __restrict__ ascale,
                      const signed char* __restrict__ w8,
                      const __half* __restrict__ wsc,
                      const __half* __restrict__ bias,
                      __half* __restrict__ out) {
    const int lane = threadIdx.x & 31;
    const int wave = threadIdx.x >> 5;
    const int lo   = lane & 15;
    const int hi   = lane >> 4;
    const int wm   = wave & 3;
    const int wn   = wave >> 2;

    const int m_wave = blockIdx.y * 128 + wm * 32;
    const int n_wave = blockIdx.x * 128 + wn * 64;

    const signed char* aptr0 = aq + (size_t)(m_wave + lo) * K_DIM + hi * 8;
    const signed char* aptr1 = aptr0 + (size_t)16 * K_DIM;
    const signed char* bptr[4];
#pragma unroll
    for (int j = 0; j < 4; ++j)
        bptr[j] = w8 + (size_t)(n_wave + j * 16 + lo) * K_DIM + hi * 8;

    v8i acc[2][4];
#pragma unroll
    for (int i = 0; i < 2; ++i)
#pragma unroll
        for (int j = 0; j < 4; ++j)
            acc[i][j] = (v8i){0, 0, 0, 0, 0, 0, 0, 0};

    for (int k0 = 0; k0 < K_DIM; k0 += 64) {
        __builtin_prefetch(aptr0 + k0 + 64, 0, 1);
        __builtin_prefetch(aptr1 + k0 + 64, 0, 1);
#pragma unroll
        for (int j = 0; j < 4; ++j)
            __builtin_prefetch(bptr[j] + k0 + 64, 0, 1);

        v8i A[2];
        {
            const signed char* p = aptr0 + k0;
            int2 x0 = *(const int2*)(p);
            int2 x1 = *(const int2*)(p + 16);
            int2 x2 = *(const int2*)(p + 32);
            int2 x3 = *(const int2*)(p + 48);
            v8i t = {x0.x, x0.y, x1.x, x1.y, x2.x, x2.y, x3.x, x3.y};
            A[0] = t;
        }
        {
            const signed char* p = aptr1 + k0;
            int2 x0 = *(const int2*)(p);
            int2 x1 = *(const int2*)(p + 16);
            int2 x2 = *(const int2*)(p + 32);
            int2 x3 = *(const int2*)(p + 48);
            v8i t = {x0.x, x0.y, x1.x, x1.y, x2.x, x2.y, x3.x, x3.y};
            A[1] = t;
        }

        v8i B[4];
#pragma unroll
        for (int j = 0; j < 4; ++j) {
            const signed char* p = bptr[j] + k0;
            int2 q0 = *(const int2*)(p);
            int2 q1 = *(const int2*)(p + 16);
            int2 q2 = *(const int2*)(p + 32);
            int2 q3 = *(const int2*)(p + 48);
            v8i t = {q0.x, q0.y, q1.x, q1.y, q2.x, q2.y, q3.x, q3.y};
            B[j] = t;
        }

#pragma unroll
        for (int i = 0; i < 2; ++i)
#pragma unroll
            for (int j = 0; j < 4; ++j)
                acc[i][j] = __builtin_amdgcn_wmma_i32_16x16x64_iu8(
                     true, A[i],  true, B[j],
                    acc[i][j],  false,  false);
    }

#pragma unroll
    for (int i = 0; i < 2; ++i) {
        const int mbase = m_wave + i * 16 + 8 * hi;
        float as[8];
#pragma unroll
        for (int r = 0; r < 8; ++r) as[r] = ascale[mbase + r];

#pragma unroll
        for (int j = 0; j < 4; ++j) {
            const int n  = n_wave + j * 16 + lo;
            const float ws = __half2float(wsc[n]);
            const float bf = __half2float(bias[n]);
#pragma unroll
            for (int r = 0; r < 8; ++r) {
                float f = (float)acc[i][j][r] * as[r] * ws + bf;
                out[(size_t)(mbase + r) * N_DIM + n] = __float2half(f);
            }
        }
    }
}

extern "C" void kernel_launch(void* const* d_in, const int* in_sizes, int n_in,
                              void* d_out, int out_size, void* d_ws, size_t ws_size,
                              hipStream_t stream) {
    const __half* x    = (const __half*)d_in[0];
    const int*    qw   = (const int*)d_in[1];
    const __half* wsc  = (const __half*)d_in[2];
    const __half* bias = (const __half*)d_in[3];
    __half*       out  = (__half*)d_out;

    const int M = in_sizes[0] / K_DIM;

    signed char* aq     = (signed char*)d_ws;
    float*       ascale = (float*)(aq + (size_t)M * K_DIM);
    signed char* w8     = (signed char*)(ascale + M + 16);

    quant_act_kernel<<<M, 256, 0, stream>>>(x, aq, ascale);

    const int total = (N_DIM * KP_DIM) / 8;
    unpack_w_kernel<<<(total + 255) / 256, 256, 0, stream>>>(qw, w8, total);

    dim3 grid(N_DIM / 128, M / 128);
    gemm_w4a4_kernel<<<grid, 256, 0, stream>>>(aq, ascale, w8, wsc, bias, out);
}
